// Cat_60730837566028
// MI455X (gfx1250) — matrix operand order repaired and checked by reading, not compiled, not run on hardware
//
#include <hip/hip_runtime.h>
#include <hip/hip_bf16.h>
#include <math.h>

typedef __attribute__((ext_vector_type(16))) _Float16 v16h;
typedef __attribute__((ext_vector_type(8)))  _Float16 v8h;
typedef __attribute__((ext_vector_type(8)))  float    v8f;

// ---------------------------------------------------------------------------
// Kernel 1: L = emb @ proj (exact f32), row max, store exp(L - max) as f16.
// One block handles 32 rows x 256 cols, K=256. blockDim = 256.
// ---------------------------------------------------------------------------
__global__ void proj_exp_kernel(const float* __restrict__ emb,
                                const float* __restrict__ proj,
                                _Float16* __restrict__ Pout,
                                float* __restrict__ amax) {
    __shared__ union SharedU {
        float rowT[256][36];   // transposed 32-row tile (padded vs bank conflicts)
        float L[32][257];      // L tile for cross-thread row-max (padded)
    } u;
    __shared__ float amaxS[32];

    const int f = threadIdx.x;          // output column 0..255
    const int row0 = blockIdx.x * 32;

    // Load 32 rows transposed: rowT[k][r] = emb[row0+r][k]; coalesced over f.
    #pragma unroll
    for (int i = 0; i < 32; ++i)
        u.rowT[f][i] = emb[(size_t)(row0 + i) * 256 + f];
    __syncthreads();

    float acc[32];
    #pragma unroll
    for (int r = 0; r < 32; ++r) acc[r] = 0.f;

    for (int k = 0; k < 256; ++k) {
        float pk = proj[k * 256 + f];                       // coalesced
        const float4* rp = (const float4*)(&u.rowT[k][0]);  // LDS broadcast
        #pragma unroll
        for (int j = 0; j < 8; ++j) {
            float4 rv = rp[j];
            acc[4*j+0] = fmaf(rv.x, pk, acc[4*j+0]);
            acc[4*j+1] = fmaf(rv.y, pk, acc[4*j+1]);
            acc[4*j+2] = fmaf(rv.z, pk, acc[4*j+2]);
            acc[4*j+3] = fmaf(rv.w, pk, acc[4*j+3]);
        }
    }
    __syncthreads();   // done reading rowT

    #pragma unroll
    for (int r = 0; r < 32; ++r) u.L[r][f] = acc[r];
    __syncthreads();

    if (f < 32) {   // per-row max across 256 columns (padded stride -> no conflicts)
        float mx = -INFINITY;
        for (int k = 0; k < 256; ++k) mx = fmaxf(mx, u.L[f][k]);
        amaxS[f] = mx;
        amax[row0 + f] = mx;
    }
    __syncthreads();

    #pragma unroll
    for (int r = 0; r < 32; ++r) {
        float v = expf(acc[r] - amaxS[r]);   // in (0, 1]: safe for f16
        Pout[(size_t)(row0 + r) * 256 + f] = (_Float16)v;
    }
}

// ---------------------------------------------------------------------------
// Kernel 2: logits = log(P @ Q^T) + a[n] + b[m], f16 WMMA, f32 accumulate.
// 128x128 tile per block, 256 threads = 8 wave32 arranged 4(n) x 2(m),
// each wave computes 32n x 64m = 2x4 tiles of v_wmma_f32_16x16x32_f16.
// ---------------------------------------------------------------------------
#define LDT 40   // padded LDS row stride in halves (80B, 16B-aligned, conflict-free)

__global__ void logits_gemm_kernel(const _Float16* __restrict__ P,
                                   const _Float16* __restrict__ Q,
                                   const float* __restrict__ a,
                                   const float* __restrict__ b,
                                   float* __restrict__ out, int Mtot) {
    __shared__ __align__(16) _Float16 As[128 * LDT];
    __shared__ __align__(16) _Float16 Bs[128 * LDT];
    __shared__ float aS[128], bS[128];

    const int t    = threadIdx.x;
    const int lane = t & 31;
    const int wave = t >> 5;
    const int wn   = wave >> 1;    // 0..3 -> 32-row n slab
    const int wm   = wave & 1;     // 0..1 -> 64-col m slab
    const int n0   = blockIdx.y * 128;
    const int m0   = blockIdx.x * 128;

    if (t < 128) { aS[t] = a[n0 + t]; bS[t] = b[m0 + t]; }

    v8f acc[2][4];
    const v8f vzero = {0.f,0.f,0.f,0.f,0.f,0.f,0.f,0.f};
    #pragma unroll
    for (int ni = 0; ni < 2; ++ni)
        #pragma unroll
        for (int mi = 0; mi < 4; ++mi) acc[ni][mi] = vzero;

    const int lr = t >> 1;           // tile row 0..127 for cooperative loads
    const int lk = (t & 1) * 16;     // half-row 0 / 16

    for (int k0 = 0; k0 < 256; k0 += 32) {
        // Cooperative 128x32 tile loads (each thread: 2 x 16B)
        {
            const v8h* gp = (const v8h*)(P + (size_t)(n0 + lr) * 256 + k0 + lk);
            v8h x0 = gp[0], x1 = gp[1];
            *(v8h*)(As + lr * LDT + lk)     = x0;
            *(v8h*)(As + lr * LDT + lk + 8) = x1;
            const v8h* gq = (const v8h*)(Q + (size_t)(m0 + lr) * 256 + k0 + lk);
            v8h y0 = gq[0], y1 = gq[1];
            *(v8h*)(Bs + lr * LDT + lk)     = y0;
            *(v8h*)(Bs + lr * LDT + lk + 8) = y1;
        }
        if (k0 + 32 < 256) {   // hint next K-tile into cache (global_prefetch_b8)
            __builtin_prefetch(P + (size_t)(n0 + lr) * 256 + k0 + 32 + lk, 0, 1);
            __builtin_prefetch(Q + (size_t)(m0 + lr) * 256 + k0 + 32 + lk, 0, 1);
        }
        __syncthreads();

        // A fragment (16x32 f16): lane<16 -> K 0-7 & 16-23, lane>=16 -> +8
        v16h af[2], bf[4];
        const int frow = lane & 15;
        const int ak   = (lane >> 4) << 3;
        #pragma unroll
        for (int ni = 0; ni < 2; ++ni) {
            const _Float16* p = As + (wn * 32 + ni * 16 + frow) * LDT + ak;
            v8h lo = *(const v8h*)p;
            v8h hi = *(const v8h*)(p + 16);
            #pragma unroll
            for (int j = 0; j < 8; ++j) { af[ni][j] = lo[j]; af[ni][j + 8] = hi[j]; }
        }
        // B fragment (32x16 f16): lane<16 -> K 0-7 & 16-23, lane>=16 -> +8
        const int bk = (lane >> 4) << 3;
        #pragma unroll
        for (int mi = 0; mi < 4; ++mi) {
            const _Float16* p = Bs + (wm * 64 + mi * 16 + frow) * LDT + bk;
            v8h lo = *(const v8h*)p;
            v8h hi = *(const v8h*)(p + 16);
            #pragma unroll
            for (int j = 0; j < 8; ++j) { bf[mi][j] = lo[j]; bf[mi][j + 8] = hi[j]; }
        }

        #pragma unroll
        for (int ni = 0; ni < 2; ++ni)
            #pragma unroll
            for (int mi = 0; mi < 4; ++mi)
                acc[ni][mi] = __builtin_amdgcn_wmma_f32_16x16x32_f16(
                    false, af[ni], false, bf[mi], (short)0, acc[ni][mi], false, false);
        __syncthreads();
    }

    // Epilogue: logits = log(S) + a[n] + b[m]; C layout: elem r -> (M = r + 8*hi, N = lane&15)
    const int en = (lane >> 4) << 3;
    const int em = lane & 15;
    #pragma unroll
    for (int ni = 0; ni < 2; ++ni) {
        #pragma unroll
        for (int mi = 0; mi < 4; ++mi) {
            const int mcol = wm * 64 + mi * 16 + em;
            const float bm = bS[mcol];
            #pragma unroll
            for (int r = 0; r < 8; ++r) {
                const int nrow = wn * 32 + ni * 16 + en + r;
                float s = fmaxf(acc[ni][mi][r], 1e-30f);   // guard log(0)
                out[(size_t)(n0 + nrow) * Mtot + (m0 + mcol)] = logf(s) + aS[nrow] + bm;
            }
        }
    }
}

// ---------------------------------------------------------------------------
// Kernel 3: per-row streaming (online) logsumexp over M columns. One block/row.
// ---------------------------------------------------------------------------
__global__ void lse_kernel(const float* __restrict__ logits,
                           float* __restrict__ lse, int Mtot) {
    const int n = blockIdx.x;
    const int t = threadIdx.x;
    const float* row = logits + (size_t)n * Mtot;
    float m = -INFINITY, s = 0.f;
    for (int i = t; i < Mtot; i += 256) {
        float x = row[i];
        if (x > m) { s = s * expf(m - x) + 1.f; m = x; }
        else       { s += expf(x - m); }
    }
    __shared__ float ms[256], ss[256];
    ms[t] = m; ss[t] = s;
    __syncthreads();
    for (int off = 128; off > 0; off >>= 1) {
        if (t < off) {
            float m2 = ms[t + off], s2 = ss[t + off];
            float mm = fmaxf(ms[t], m2);
            ss[t] = ss[t] * expf(ms[t] - mm) + s2 * expf(m2 - mm);
            ms[t] = mm;
        }
        __syncthreads();
    }
    if (t == 0) lse[n] = ms[0] + logf(ss[0]);
}

// ---------------------------------------------------------------------------
// Kernel 4: out[n, m] -= lse[n], vectorized float4.
// ---------------------------------------------------------------------------
__global__ void sub_kernel(float* __restrict__ out, const float* __restrict__ lse,
                           int mlog2, size_t total) {
    size_t idx = ((size_t)blockIdx.x * blockDim.x + threadIdx.x) * 4;
    if (idx >= total) return;
    float l = lse[idx >> mlog2];
    float4* p = (float4*)(out + idx);
    float4 v = *p;
    v.x -= l; v.y -= l; v.z -= l; v.w -= l;
    *p = v;
}

// ---------------------------------------------------------------------------
extern "C" void kernel_launch(void* const* d_in, const int* in_sizes, int n_in,
                              void* d_out, int out_size, void* d_ws, size_t ws_size,
                              hipStream_t stream) {
    const float* start_emb  = (const float*)d_in[0];   // [N, 256] f32
    const float* output_emb = (const float*)d_in[1];   // [M, 256] f32
    const float* proj       = (const float*)d_in[2];   // [256, 256] f32
    float* out = (float*)d_out;

    const int E = 256;
    const int N = in_sizes[0] / E;   // 4096
    const int M = in_sizes[1] / E;   // 32768

    // Workspace layout (all 256B aligned):
    char* ws = (char*)d_ws;
    size_t off = 0;
    _Float16* P = (_Float16*)(ws + off); off += (size_t)N * E * 2; off = (off + 255) & ~(size_t)255;
    _Float16* Q = (_Float16*)(ws + off); off += (size_t)M * E * 2; off = (off + 255) & ~(size_t)255;
    float*    a = (float*)(ws + off);    off += (size_t)N * 4;     off = (off + 255) & ~(size_t)255;
    float*    b = (float*)(ws + off);    off += (size_t)M * 4;     off = (off + 255) & ~(size_t)255;
    float*  lse = (float*)(ws + off);

    proj_exp_kernel<<<N / 32, 256, 0, stream>>>(start_emb,  proj, P, a);
    proj_exp_kernel<<<M / 32, 256, 0, stream>>>(output_emb, proj, Q, b);

    dim3 grid(M / 128, N / 128);
    logits_gemm_kernel<<<grid, 256, 0, stream>>>(P, Q, a, b, out, M);

    lse_kernel<<<N, 256, 0, stream>>>(out, lse, M);

    int mlog2 = 0; { int mm = M; while (mm > 1) { mm >>= 1; ++mlog2; } }
    size_t total = (size_t)N * M;
    unsigned blocks = (unsigned)((total / 4 + 255) / 256);
    sub_kernel<<<blocks, 256, 0, stream>>>(out, lse, mlog2, total);
}
